// CorticalColumn_17910013624699
// MI455X (gfx1250) — matrix operand order repaired and checked by reading, not compiled, not run on hardware
//
#include <hip/hip_runtime.h>
#include <math.h>

// ---------------------------------------------------------------------------
// CorticalColumn forward on MI455X (gfx1250, wave32, WMMA bf16)
//   B=16384, D_SEM=2048, D_MINI=256, basal S=4, apical S=3
// GEMM: block tile 128x128, 8 waves, wave tile 32x64 (8 WMMA/K-step),
// double-buffered LDS ping-pong (1 barrier per K-step), software-pipelined
// global->reg->LDS staging, N-fastest grid for L2 blocking, non-temporal
// stores for the write-once 536MB projection outputs.
// ---------------------------------------------------------------------------

typedef __attribute__((ext_vector_type(16))) __bf16 v16bf;
typedef __attribute__((ext_vector_type(8)))  float  v8f;

#define BATCH   16384
#define DSEM    2048
#define DMINI   256

#define BM 128
#define BN 128
#define BK 32
#define LDA 36   // LDS row stride (bf16): 72B = 18 banks, odd*2 -> conflict-free frag reads
#define LDB 36

union FragBF { v16bf v; __bf16 e[16]; };
union FragF  { v8f   v; float  f[8]; };

struct ARegs { float4 f[4]; };
struct BRegs { float2 f[8]; };

__device__ __forceinline__ float sigm(float x) { return 1.0f / (1.0f + expf(-x)); }

// ---- fragment builders (ISA 16-bit WMMA VGPR layouts) ----------------------
// A 16x32: lane<16 -> M=lane, K {0..7,16..23}; lane>=16 -> M=lane-16, K {8..15,24..31}
__device__ __forceinline__ v16bf frag_a16(const __bf16* base, int lane) {
  FragBF fa;
  const int row = lane & 15;
  const int kh  = (lane >> 4) << 3;
  const __bf16* p = base + row * LDA;
#pragma unroll
  for (int v = 0; v < 8; ++v) {
    const int k = ((v & 4) ? 16 : 0) + kh + ((v & 3) << 1);
    fa.e[2 * v]     = p[k];
    fa.e[2 * v + 1] = p[k + 1];
  }
  return fa.v;
}
// B 32x16 (stored transposed in LDS): lane<16 -> N=lane, K {0..7,16..23}; lane>=16 -> N=lane-16, K {8..15,24..31}
__device__ __forceinline__ v16bf frag_b16(const __bf16* baseT, int lane) {
  FragBF fb;
  const int col = lane & 15;
  const int kh  = (lane >> 4) << 3;
  const __bf16* p = baseT + col * LDB + kh;
#pragma unroll
  for (int v = 0; v < 16; ++v) fb.e[v] = p[((v & 8) ? 16 : 0) + (v & 7)];
  return fb.v;
}

// ---- two-phase tile staging ------------------------------------------------
// A tile 128x32 fp32: 1024 float4, 4 per thread.
__device__ __forceinline__ void load_a_regs(const float* __restrict__ src, int ld,
                                            int tid, ARegs& ar) {
#pragma unroll
  for (int i = 0; i < 4; ++i) {
    const int e  = tid + (i << 8);
    const int r  = e >> 3;
    const int c4 = (e & 7) << 2;
    ar.f[i] = *reinterpret_cast<const float4*>(src + (size_t)r * ld + c4);
  }
}
__device__ __forceinline__ void store_a_lds(const ARegs& ar, __bf16* As, int tid) {
#pragma unroll
  for (int i = 0; i < 4; ++i) {
    const int e  = tid + (i << 8);
    const int r  = e >> 3;
    const int c4 = (e & 7) << 2;
    __bf16* dst = As + r * LDA + c4;   // 8B-aligned -> one ds_store_b64, conflict-free
    dst[0] = (__bf16)ar.f[i].x; dst[1] = (__bf16)ar.f[i].y;
    dst[2] = (__bf16)ar.f[i].z; dst[3] = (__bf16)ar.f[i].w;
  }
}
// B tile 32x128 from row-major W (stride ldw): each thread owns a column pair
// x 8 contiguous K -> 8 coalesced float2 loads, 2 contiguous 16B LDS stores.
__device__ __forceinline__ void load_b_regs(const float* __restrict__ w, size_t ldw,
                                            int tid, BRegs& br) {
  const int colp = (tid & 63) << 1;
  const int kg   = (tid >> 6) << 3;
#pragma unroll
  for (int i = 0; i < 8; ++i)
    br.f[i] = *reinterpret_cast<const float2*>(w + (size_t)(kg + i) * ldw + colp);
}
template <int S>
__device__ __forceinline__ void load_b_gate_regs(const float* __restrict__ gw, int k0,
                                                 int tid, BRegs& br) {
  const int colp = (tid & 63) << 1;
  const int kg   = (tid >> 6) << 3;
#pragma unroll
  for (int i = 0; i < 8; ++i) {
    const int k = k0 + kg + i;
    br.f[i].x = (colp     < S) ? gw[(size_t)k * S + colp]     : 0.0f;
    br.f[i].y = (colp + 1 < S) ? gw[(size_t)k * S + colp + 1] : 0.0f;
  }
}
__device__ __forceinline__ void store_b_lds(const BRegs& br, __bf16* BsT, int tid) {
  const int colp = (tid & 63) << 1;
  const int kg   = (tid >> 6) << 3;
  __bf16* d0 = BsT + colp * LDB + kg;
  __bf16* d1 = d0 + LDB;
#pragma unroll
  for (int i = 0; i < 8; ++i) { d0[i] = (__bf16)br.f[i].x; d1[i] = (__bf16)br.f[i].y; }
}

// ---- per-wave compute: 32(M) x 64(N), 8 WMMAs ------------------------------
__device__ __forceinline__ void wmma_step(const __bf16* As, const __bf16* BsT,
                                          int wm, int wn, int lane, FragF acc[2][4]) {
  const v16bf a0 = frag_a16(As + (wm * 32) * LDA, lane);
  const v16bf a1 = frag_a16(As + (wm * 32 + 16) * LDA, lane);
  v16bf b[4];
#pragma unroll
  for (int ni = 0; ni < 4; ++ni)
    b[ni] = frag_b16(BsT + (wn * 64 + ni * 16) * LDB, lane);
#pragma unroll
  for (int ni = 0; ni < 4; ++ni) {
    acc[0][ni].v = __builtin_amdgcn_wmma_f32_16x16x32_bf16(false, a0, false, b[ni], (short)0, acc[0][ni].v, false, false);
    acc[1][ni].v = __builtin_amdgcn_wmma_f32_16x16x32_bf16(false, a1, false, b[ni], (short)0, acc[1][ni].v, false, false);
  }
}

// ---- double-buffered (ping-pong) main loop: one barrier per K-step ---------
template <typename LA, typename LB>
__device__ __forceinline__ void gemm_loop(
    int KTOT, LA&& loadA, LB&& loadB,
    __bf16 (&As)[2][BM * LDA], __bf16 (&BsT)[2][BN * LDB],
    int tid, int wm, int wn, int lane, FragF acc[2][4]) {
  ARegs ar; BRegs br;
  loadA(0, ar); loadB(0, br);
  store_a_lds(ar, As[0], tid); store_b_lds(br, BsT[0], tid);
  __syncthreads();
  for (int k0 = 0; k0 < KTOT; k0 += 2 * BK) {     // KTOT is a multiple of 64
    // stage 0: compute buf0, prefetch k0+BK into buf1
    loadA(k0 + BK, ar); loadB(k0 + BK, br);
    wmma_step(As[0], BsT[0], wm, wn, lane, acc);
    store_a_lds(ar, As[1], tid); store_b_lds(br, BsT[1], tid);
    __syncthreads();
    // stage 1: compute buf1, prefetch k0+2BK into buf0
    const bool more = (k0 + 2 * BK < KTOT);
    if (more) { loadA(k0 + 2 * BK, ar); loadB(k0 + 2 * BK, br); }
    wmma_step(As[1], BsT[1], wm, wn, lane, acc);
    if (more) {
      store_a_lds(ar, As[0], tid); store_b_lds(br, BsT[0], tid);
    }
    __syncthreads();
  }
}

#define GEMM_PROLOGUE()                                   \
  __shared__ __bf16 As[2][BM * LDA];                      \
  __shared__ __bf16 BsT[2][BN * LDB];                     \
  const int tid  = threadIdx.x;                           \
  const int lane = tid & 31;                              \
  const int wave = tid >> 5;                              \
  const int wm   = wave & 3;                              \
  const int wn   = wave >> 2;                             \
  const int nBase = blockIdx.x * BN;                      \
  const int mBase = blockIdx.y * BM;                      \
  FragF acc[2][4];                                        \
  _Pragma("unroll") for (int a_ = 0; a_ < 2; ++a_)        \
  _Pragma("unroll") for (int b_ = 0; b_ < 4; ++b_)        \
  _Pragma("unroll") for (int r_ = 0; r_ < 8; ++r_) acc[a_][b_].f[r_] = 0.0f;

// ---------------------------------------------------------------------------
// Kernel 1: dendritic GEMM. seg = X @ W(K x S*256); gates (X @ gw) folded in
// as one extra N-tile. Non-gate tiles lie fully inside one weight segment.
// ---------------------------------------------------------------------------
template <int S>
__global__ __launch_bounds__(256) void dendritic_gemm(
    const float* __restrict__ X,      // BATCH x DSEM
    const float* __restrict__ W,      // S x DSEM x DMINI
    const float* __restrict__ gw,     // DSEM x S
    float* __restrict__ seg,          // BATCH x (S*DMINI)
    float* __restrict__ gates) {      // BATCH x S (raw)
  constexpr int NS = S * DMINI;
  GEMM_PROLOGUE();
  const bool gateTile = (nBase >= NS);
  const float* wptr = W + (size_t)(nBase >> 8) * DSEM * DMINI + (nBase & 255);
  const float* aptr = X + (size_t)mBase * DSEM;

  auto loadA = [&](int k, ARegs& ar) { load_a_regs(aptr + k, DSEM, tid, ar); };
  if (!gateTile) {
    auto loadB = [&](int k, BRegs& br) { load_b_regs(wptr + (size_t)k * DMINI, DMINI, tid, br); };
    gemm_loop(DSEM, loadA, loadB, As, BsT, tid, wm, wn, lane, acc);
  } else {
    auto loadB = [&](int k, BRegs& br) { load_b_gate_regs<S>(gw, k, tid, br); };
    gemm_loop(DSEM, loadA, loadB, As, BsT, tid, wm, wn, lane, acc);
  }

  const int colL   = lane & 15;
  const int rowOff = (lane >> 4) << 3;
  if (!gateTile) {
#pragma unroll
    for (int mi = 0; mi < 2; ++mi)
#pragma unroll
      for (int ni = 0; ni < 4; ++ni) {
        const int gcol = nBase + wn * 64 + ni * 16 + colL;
#pragma unroll
        for (int r = 0; r < 8; ++r) {
          const int grow = mBase + wm * 32 + mi * 16 + rowOff + r;
          seg[(size_t)grow * NS + gcol] = acc[mi][ni].f[r];
        }
      }
  } else {
    const int gc = wn * 64 + colL;                // gate column (ni=0 subtile)
    if (gc < S) {
#pragma unroll
      for (int mi = 0; mi < 2; ++mi)
#pragma unroll
        for (int r = 0; r < 8; ++r) {
          const int grow = mBase + wm * 32 + mi * 16 + rowOff + r;
          gates[(size_t)grow * S + gc] = acc[mi][0].f[r];
        }
    }
  }
}

// ---------------------------------------------------------------------------
// Kernel 2 (fused): dendritic nonlinearity for both compartments + pred_error
// + burst. One block per batch row (256 threads = 256 minicolumn channels).
// ---------------------------------------------------------------------------
template <int S>
__device__ __forceinline__ float dend_combine(
    const float* __restrict__ seg, const float* __restrict__ gates_raw,
    const float* __restrict__ bseg, const float* __restrict__ thr,
    const float* __restrict__ gb, int b, int o) {
  float total = 0.0f, prod = 1.0f, pabs = 1.0f;
#pragma unroll
  for (int s = 0; s < S; ++s) {
    const float g  = sigm(gates_raw[(size_t)b * S + s] + gb[s]);
    const float sv = seg[(size_t)b * S * DMINI + s * DMINI + o] + bseg[s * DMINI + o];
    const float pl = sigm(5.0f * (sv - thr[s * DMINI + o]));
    const float gt = sv * pl * g;
    total += gt; prod *= gt; pabs *= fabsf(gt);
  }
  const float mag = (S == 4) ? sqrtf(sqrtf(pabs))
                             : exp2f(log2f(pabs) * (1.0f / (float)S));
  const float sgn = (prod > 0.0f) ? 1.0f : ((prod < 0.0f) ? -1.0f : 0.0f);
  return total + 0.1f * sgn * mag;
}

__global__ __launch_bounds__(256) void neuron_fused_kernel(
    const float* __restrict__ seg_b, const float* __restrict__ gat_b,
    const float* __restrict__ basal_bi, const float* __restrict__ basal_thr,
    const float* __restrict__ basal_gb,
    const float* __restrict__ seg_a, const float* __restrict__ gat_a,
    const float* __restrict__ apical_bi, const float* __restrict__ apical_thr,
    const float* __restrict__ apical_gb,
    const float* __restrict__ bw, const float* __restrict__ bb,
    float* __restrict__ basalv, float* __restrict__ apicalv,
    float* __restrict__ predv, float* __restrict__ burst_out) {
  __shared__ float red[8];
  const int b = blockIdx.x, o = threadIdx.x;
  const int lane = o & 31, wave = o >> 5;
  const float bas = dend_combine<4>(seg_b, gat_b, basal_bi, basal_thr, basal_gb, b, o);
  const float api = dend_combine<3>(seg_a, gat_a, apical_bi, apical_thr, apical_gb, b, o);
  const size_t idx = (size_t)b * DMINI + o;
  basalv[idx] = bas;
  apicalv[idx] = api;
  const float pe = bas - api;
  predv[idx] = pe;
  float s = fabsf(pe) * bw[o];
#pragma unroll
  for (int off = 16; off; off >>= 1) s += __shfl_xor(s, off, 32);
  if (lane == 0) red[wave] = s;
  __syncthreads();
  if (o == 0) {
    float t = 0.0f;
#pragma unroll
    for (int w = 0; w < 8; ++w) t += red[w];
    burst_out[b] = sigm(t + bb[0]);
  }
}

// ---------------------------------------------------------------------------
// Kernel 3: soma GEMM (K=512 from basal||apical, N=256) + exact GELU,
// with the burst amplification fused into the epilogue.
// ---------------------------------------------------------------------------
__global__ __launch_bounds__(256) void soma_gemm(
    const float* __restrict__ basal, const float* __restrict__ apical,
    const float* __restrict__ Wm,    // 512 x 256 row-major
    const float* __restrict__ bias, const float* __restrict__ burst,
    float* __restrict__ somav, float* __restrict__ amplified) {
  GEMM_PROLOGUE();
  const float* wptr = Wm + nBase;

  auto loadA = [&](int k, ARegs& ar) {
    const float* src = (k < DMINI) ? basal : apical;
    const int koff   = (k < DMINI) ? k : (k - DMINI);
    load_a_regs(src + (size_t)mBase * DMINI + koff, DMINI, tid, ar);
  };
  auto loadB = [&](int k, BRegs& br) { load_b_regs(wptr + (size_t)k * DMINI, DMINI, tid, br); };
  gemm_loop(2 * DMINI, loadA, loadB, As, BsT, tid, wm, wn, lane, acc);

  const int colL = lane & 15, rowOff = (lane >> 4) << 3;
#pragma unroll
  for (int mi = 0; mi < 2; ++mi)
#pragma unroll
    for (int ni = 0; ni < 4; ++ni) {
      const int gcol = nBase + wn * 64 + ni * 16 + colL;
#pragma unroll
      for (int r = 0; r < 8; ++r) {
        const int grow = mBase + wm * 32 + mi * 16 + rowOff + r;
        float y = acc[mi][ni].f[r] + bias[gcol];
        y = 0.5f * y * (1.0f + erff(y * 0.70710678118654752f));   // exact GELU
        somav[(size_t)grow * DMINI + gcol] = y;
        amplified[(size_t)grow * DMINI + gcol] = y * (1.0f + 2.0f * burst[grow]);
      }
    }
}

// ---------------------------------------------------------------------------
// Kernel 4: output projections (BATCH x 256) @ (256 x 2048) + bias.
// Outputs are write-once streaming data -> non-temporal stores keep the
// L2-resident activations/weights from being evicted.
// ---------------------------------------------------------------------------
__global__ __launch_bounds__(256) void proj_gemm(
    const float* __restrict__ A,      // BATCH x DMINI
    const float* __restrict__ Wp,     // DMINI x DSEM row-major
    const float* __restrict__ bias, float* __restrict__ outp) {
  GEMM_PROLOGUE();
  const float* wptr = Wp + nBase;
  const float* aptr = A + (size_t)mBase * DMINI;

  auto loadA = [&](int k, ARegs& ar) { load_a_regs(aptr + k, DMINI, tid, ar); };
  auto loadB = [&](int k, BRegs& br) { load_b_regs(wptr + (size_t)k * DSEM, DSEM, tid, br); };
  gemm_loop(DMINI, loadA, loadB, As, BsT, tid, wm, wn, lane, acc);

  const int colL = lane & 15, rowOff = (lane >> 4) << 3;
#pragma unroll
  for (int mi = 0; mi < 2; ++mi)
#pragma unroll
    for (int ni = 0; ni < 4; ++ni) {
      const int gcol = nBase + wn * 64 + ni * 16 + colL;
#pragma unroll
      for (int r = 0; r < 8; ++r) {
        const int grow = mBase + wm * 32 + mi * 16 + rowOff + r;
        __builtin_nontemporal_store(acc[mi][ni].f[r] + bias[gcol],
                                    outp + (size_t)grow * DSEM + gcol);
      }
    }
}

// ---------------------------------------------------------------------------
extern "C" void kernel_launch(void* const* d_in, const int* in_sizes, int n_in,
                              void* d_out, int out_size, void* d_ws, size_t ws_size,
                              hipStream_t stream) {
  (void)in_sizes; (void)n_in; (void)out_size; (void)ws_size;
  const float* bottom_up = (const float*)d_in[0];
  const float* top_down  = (const float*)d_in[1];
  const float* basal_w   = (const float*)d_in[2];
  const float* basal_bi  = (const float*)d_in[3];
  const float* basal_thr = (const float*)d_in[4];
  const float* basal_gw  = (const float*)d_in[5];
  const float* basal_gb  = (const float*)d_in[6];
  const float* apical_w  = (const float*)d_in[7];
  const float* apical_bi = (const float*)d_in[8];
  const float* apical_thr= (const float*)d_in[9];
  const float* apical_gw = (const float*)d_in[10];
  const float* apical_gb = (const float*)d_in[11];
  const float* soma_w    = (const float*)d_in[12];
  const float* soma_b    = (const float*)d_in[13];
  const float* burst_w   = (const float*)d_in[14];
  const float* burst_b   = (const float*)d_in[15];
  const float* l23_w = (const float*)d_in[16];
  const float* l23_b = (const float*)d_in[17];
  const float* l5_w  = (const float*)d_in[18];
  const float* l5_b  = (const float*)d_in[19];
  const float* l6_w  = (const float*)d_in[20];
  const float* l6_b  = (const float*)d_in[21];
  const float* err_w = (const float*)d_in[22];
  const float* err_b = (const float*)d_in[23];

  // Output layout (concat, return order): layer23, layer5, layer6, error, burst, amplified
  float* out      = (float*)d_out;
  float* layer23  = out;
  float* layer5   = layer23 + (size_t)BATCH * DSEM;
  float* layer6   = layer5  + (size_t)BATCH * DSEM;
  float* errorp   = layer6  + (size_t)BATCH * DSEM;
  float* burst    = errorp  + (size_t)BATCH * DSEM;
  float* amplified= burst   + (size_t)BATCH;

  // Workspace (~185 MB fp32)
  float* ws      = (float*)d_ws;
  float* seg_b   = ws;                                  // BATCH * 1024
  float* seg_a   = seg_b  + (size_t)BATCH * 4 * DMINI;  // BATCH * 768
  float* gat_b   = seg_a  + (size_t)BATCH * 3 * DMINI;  // BATCH * 4
  float* gat_a   = gat_b  + (size_t)BATCH * 4;          // BATCH * 3 (pad 4)
  float* basalv  = gat_a  + (size_t)BATCH * 4;          // BATCH * 256
  float* apicalv = basalv + (size_t)BATCH * DMINI;
  float* predv   = apicalv+ (size_t)BATCH * DMINI;
  float* somav   = predv  + (size_t)BATCH * DMINI;

  const dim3 blk(256);
  const int mTiles = BATCH / BM;                        // 128

  // 1. Dendritic GEMMs (N-fastest grid: all N-blocks of an M-tile share X in L2)
  dendritic_gemm<4><<<dim3(4 * DMINI / BN + 1, mTiles), blk, 0, stream>>>(
      bottom_up, basal_w, basal_gw, seg_b, gat_b);
  dendritic_gemm<3><<<dim3(3 * DMINI / BN + 1, mTiles), blk, 0, stream>>>(
      top_down, apical_w, apical_gw, seg_a, gat_a);

  // 2. Fused dendritic nonlinearity + pred_error + burst
  neuron_fused_kernel<<<BATCH, blk, 0, stream>>>(
      seg_b, gat_b, basal_bi, basal_thr, basal_gb,
      seg_a, gat_a, apical_bi, apical_thr, apical_gb,
      burst_w, burst_b, basalv, apicalv, predv, burst);

  // 3. Soma GEMM + GELU + fused amplification
  soma_gemm<<<dim3(2 * DMINI / BN, mTiles), blk, 0, stream>>>(
      basalv, apicalv, soma_w, soma_b, burst, somav, amplified);

  // 4. Output projections (non-temporal stores)
  proj_gemm<<<dim3(DSEM / BN, mTiles), blk, 0, stream>>>(amplified, l23_w, l23_b, layer23);
  proj_gemm<<<dim3(DSEM / BN, mTiles), blk, 0, stream>>>(amplified, l5_w,  l5_b,  layer5);
  proj_gemm<<<dim3(DSEM / BN, mTiles), blk, 0, stream>>>(somav,     l6_w,  l6_b,  layer6);
  proj_gemm<<<dim3(DSEM / BN, mTiles), blk, 0, stream>>>(predv,     err_w, err_b, errorp);
}
